// Model_79972291051897
// MI455X (gfx1250) — matrix operand order repaired and checked by reading, not compiled, not run on hardware
//
#include <hip/hip_runtime.h>
#include <hip/hip_bf16.h>
#include <math.h>

typedef __attribute__((ext_vector_type(16))) __bf16 v16bf;
typedef __attribute__((ext_vector_type(8)))  float  v8f;

#define N_NODES 50000
#define N_EDGES 512000
#define N_GRAPHS 64
#define CDIM 64
#define HC 128   // H * C

// ---------------------------------------------------------------------------
// WMMA GEMM: Y[M,NOUT] = A[M,K] @ W[K,NOUT] + bias,  f32 in/out, bf16 WMMA core.
// KP = padded K (32 or 64), NOUT in {64,128}. blockDim.x = 2*NOUT (NOUT/16 waves).
// Each block computes a 64-row M tile (4 subtiles of 16) across all NOUT columns.
// Per k-step each wave loads its B fragment once and reuses it for 4 WMMAs.
// Requires M % 16 == 0 (true here: 50000, 512000) so subtile store guards are
// uniform scalar branches (no per-lane exec divergence in the epilogue).
// ---------------------------------------------------------------------------
template <int KP, int NOUT>
__global__ void gemm_bias_wmma(const float* __restrict__ A,
                               const float* __restrict__ W,
                               const float* __restrict__ bias,
                               float* __restrict__ Y,
                               int M, int K) {
    __shared__ __bf16 sA[64 * KP];        // A tile, row-major [64][KP]
    __shared__ __bf16 sBt[NOUT * KP];     // W transposed    [NOUT][KP]

    const int t   = threadIdx.x;
    const int nth = blockDim.x;
    const int row0 = blockIdx.x * 64;

    // Stage A tile (f32 -> bf16, zero pad K..KP and rows >= M)
    for (int idx = t; idx < 64 * KP; idx += nth) {
        int r = idx / KP, kk = idx % KP;
        int gr = row0 + r;
        float val = (gr < M && kk < K) ? A[(size_t)gr * K + kk] : 0.0f;
        sA[idx] = (__bf16)val;
    }
    // Stage W transposed (f32 -> bf16, zero pad)
    for (int idx = t; idx < NOUT * KP; idx += nth) {
        int n = idx / KP, kk = idx % KP;
        float val = (kk < K) ? W[(size_t)kk * NOUT + n] : 0.0f;
        sBt[idx] = (__bf16)val;
    }
    // Prefetch next block's A tile into cache (global_prefetch path).
    if (row0 + 64 < M) {
        const char* nxtA = (const char*)(A + (size_t)(row0 + 64) * K);
        const int bytes = 64 * K * 4;
        if (t * 128 < bytes) __builtin_prefetch(nxtA + t * 128, 0, 1);
    }
    __syncthreads();

    const int wave = t >> 5;
    const int lane = t & 31;
    const int m    = lane & 15;       // row within subtile / column within wave tile
    const int half = lane >> 4;       // lane group 0/1
    const int ncol0 = wave * 16;

    v8f acc0 = {}, acc1 = {}, acc2 = {}, acc3 = {};
#pragma unroll
    for (int k0 = 0; k0 < KP; k0 += 32) {
        // B fragment: lanes 0-15: K = k0+{0..7,16..23}; lanes 16-31: +8, for column ncol0+m.
        // Loaded once per k-step, reused across the 4 m-subtiles.
        v16bf b;
        const __bf16* bp = &sBt[(ncol0 + m) * KP + k0 + half * 8];
#pragma unroll
        for (int j = 0; j < 8; ++j) { b[j] = bp[j]; b[j + 8] = bp[j + 16]; }

        // A fragments: lanes 0-15: K = k0+{0..7,16..23}; lanes 16-31: +8
        const __bf16* ap = &sA[m * KP + k0 + half * 8];
        v16bf a;
#pragma unroll
        for (int j = 0; j < 8; ++j) { a[j] = ap[j]; a[j + 8] = ap[j + 16]; }
        acc0 = __builtin_amdgcn_wmma_f32_16x16x32_bf16(false, a, false, b, (short)0, acc0, false, false);

#pragma unroll
        for (int j = 0; j < 8; ++j) { a[j] = ap[16 * KP + j]; a[j + 8] = ap[16 * KP + j + 16]; }
        acc1 = __builtin_amdgcn_wmma_f32_16x16x32_bf16(false, a, false, b, (short)0, acc1, false, false);

#pragma unroll
        for (int j = 0; j < 8; ++j) { a[j] = ap[32 * KP + j]; a[j + 8] = ap[32 * KP + j + 16]; }
        acc2 = __builtin_amdgcn_wmma_f32_16x16x32_bf16(false, a, false, b, (short)0, acc2, false, false);

#pragma unroll
        for (int j = 0; j < 8; ++j) { a[j] = ap[48 * KP + j]; a[j + 8] = ap[48 * KP + j + 16]; }
        acc3 = __builtin_amdgcn_wmma_f32_16x16x32_bf16(false, a, false, b, (short)0, acc3, false, false);
    }

    // D layout: VGPR r <-> row (subtile*16 + r + 8*half), col = ncol0 + m.
    // Uniform (scalar) per-subtile guards: M % 16 == 0.
    const int col = ncol0 + m;
    const float bv = bias[col];
    {
        v8f accs[4] = {acc0, acc1, acc2, acc3};
#pragma unroll
        for (int mt = 0; mt < 4; ++mt) {
            if (row0 + mt * 16 < M) {
                float* yp = Y + (size_t)(row0 + mt * 16 + 8 * half) * NOUT + col;
#pragma unroll
                for (int r = 0; r < 8; ++r) yp[(size_t)r * NOUT] = accs[mt][r] + bv;
            }
        }
    }
}

// ---------------------------------------------------------------------------
// Utility kernels
// ---------------------------------------------------------------------------
__global__ void fill_kernel(float* __restrict__ p, float v, long n) {
    long i = (long)blockIdx.x * blockDim.x + threadIdx.x;
    if (i < n) p[i] = v;
}

__device__ inline void atomicMaxFloat(float* addr, float val) {
    int* ai = (int*)addr;
    int old = __float_as_int(*addr);
    while (__int_as_float(old) < val) {
        int assumed = old;
        old = atomicCAS(ai, assumed, __float_as_int(val));
        if (old == assumed) break;
    }
}

// logits[e][h] = scale * dot(q[dst[e]][h], k[src[e]][h] + ep[e][h])
__global__ void edge_logits(const float* __restrict__ q, const float* __restrict__ k,
                            const float* __restrict__ ep,
                            const int* __restrict__ src, const int* __restrict__ dst,
                            float* __restrict__ lg, int E) {
    int e = blockIdx.x * blockDim.x + threadIdx.x;
    if (e >= E) return;
    const int s = src[e], d = dst[e];
    const float scale = 0.125f; // 1/sqrt(64)
#pragma unroll
    for (int hh = 0; hh < 2; ++hh) {
        const float* qp  = q  + (size_t)d * HC + hh * CDIM;
        const float* kp  = k  + (size_t)s * HC + hh * CDIM;
        const float* epp = ep + (size_t)e * HC + hh * CDIM;
        float acc = 0.0f;
#pragma unroll
        for (int c = 0; c < CDIM; c += 4) {
            float4 qv = *(const float4*)(qp + c);
            float4 kv = *(const float4*)(kp + c);
            float4 ev = *(const float4*)(epp + c);
            acc += qv.x * (kv.x + ev.x) + qv.y * (kv.y + ev.y) +
                   qv.z * (kv.z + ev.z) + qv.w * (kv.w + ev.w);
        }
        lg[(size_t)e * 2 + hh] = acc * scale;
    }
}

__global__ void seg_max(const float* __restrict__ lg, const int* __restrict__ dst,
                        float* __restrict__ m, int E) {
    long i = (long)blockIdx.x * blockDim.x + threadIdx.x;
    if (i >= (long)E * 2) return;
    int e = (int)(i >> 1), hh = (int)(i & 1);
    atomicMaxFloat(&m[(size_t)dst[e] * 2 + hh], lg[i]);
}

__global__ void seg_expsum(float* __restrict__ lg, const int* __restrict__ dst,
                           const float* __restrict__ m, float* __restrict__ dn, int E) {
    long i = (long)blockIdx.x * blockDim.x + threadIdx.x;
    if (i >= (long)E * 2) return;
    int e = (int)(i >> 1), hh = (int)(i & 1);
    int d = dst[e];
    float a = __expf(lg[i] - m[(size_t)d * 2 + hh]);
    lg[i] = a;
    atomicAdd(&dn[(size_t)d * 2 + hh], a);
}

// nxt[dst][c] += 0.5 * alpha[e][h] * (v[src][h][c] + ep[e][h][c])   (mean over heads)
__global__ void edge_agg(const float* __restrict__ lg, const float* __restrict__ dn,
                         const float* __restrict__ v, const float* __restrict__ ep,
                         const int* __restrict__ src, const int* __restrict__ dst,
                         float* __restrict__ nxt, int E) {
    long i = (long)blockIdx.x * blockDim.x + threadIdx.x;
    if (i >= (long)E * HC) return;
    int e = (int)(i >> 7);
    int j = (int)(i & 127);
    int hh = j >> 6, c = j & 63;
    int s = src[e], d = dst[e];
    float alpha = lg[(size_t)e * 2 + hh] / dn[(size_t)d * 2 + hh];
    float val = 0.5f * alpha * (v[(size_t)s * HC + j] + ep[(size_t)e * HC + j]);
    atomicAdd(&nxt[(size_t)d * CDIM + c], val);
}

__global__ void bn_stats(const float* __restrict__ x, float* __restrict__ ssum,
                         float* __restrict__ ssq, int N) {
    __shared__ float ls[CDIM], lq[CDIM];
    int t = threadIdx.x;
    if (t < CDIM) { ls[t] = 0.0f; lq[t] = 0.0f; }
    __syncthreads();
    int c = t & 63;
    int rowsPerIter = blockDim.x >> 6;
    float s = 0.0f, q = 0.0f;
    for (int r = blockIdx.x * rowsPerIter + (t >> 6); r < N; r += gridDim.x * rowsPerIter) {
        float vv = x[(size_t)r * CDIM + c];
        s += vv; q += vv * vv;
    }
    atomicAdd(&ls[c], s);
    atomicAdd(&lq[c], q);
    __syncthreads();
    if (t < CDIM) { atomicAdd(&ssum[t], ls[t]); atomicAdd(&ssq[t], lq[t]); }
}

__global__ void bn_finalize(const float* __restrict__ ssum, const float* __restrict__ ssq,
                            float* __restrict__ mu, float* __restrict__ rstd, float invN) {
    int c = threadIdx.x;
    if (c < CDIM) {
        float m_ = ssum[c] * invN;
        float var = ssq[c] * invN - m_ * m_;
        mu[c] = m_;
        rstd[c] = rsqrtf(var + 1e-5f);
    }
}

// h = leaky_relu(gamma*(nxt-mu)*rstd + beta) + h   (residual fused)
__global__ void bn_apply(float* __restrict__ h, const float* __restrict__ nxt,
                         const float* __restrict__ mu, const float* __restrict__ rstd,
                         const float* __restrict__ gamma, const float* __restrict__ beta,
                         int N) {
    long i = (long)blockIdx.x * blockDim.x + threadIdx.x;
    if (i >= (long)N * CDIM) return;
    int c = (int)(i & 63);
    float o = gamma[c] * (nxt[i] - mu[c]) * rstd[c] + beta[c];
    o = (o > 0.0f) ? o : 0.01f * o;
    h[i] = h[i] + o;
}

__global__ void gate_kernel(const float* __restrict__ h, const float* __restrict__ gW,
                            const float* __restrict__ gb, float* __restrict__ gate, int N) {
    int n = blockIdx.x * blockDim.x + threadIdx.x;
    if (n >= N) return;
    const float* hp = h + (size_t)n * CDIM;
    float acc = gb[0];
#pragma unroll
    for (int c = 0; c < CDIM; c += 4) {
        float4 hv = *(const float4*)(hp + c);
        float4 wv = *(const float4*)(gW + c);
        acc += hv.x * wv.x + hv.y * wv.y + hv.z * wv.z + hv.w * wv.w;
    }
    gate[n] = acc;
}

__global__ void pool_max(const float* __restrict__ gate, const int* __restrict__ batch,
                         float* __restrict__ gm, int N) {
    int n = blockIdx.x * blockDim.x + threadIdx.x;
    if (n >= N) return;
    atomicMaxFloat(&gm[batch[n]], gate[n]);
}

__global__ void pool_expsum(float* __restrict__ gate, const int* __restrict__ batch,
                            const float* __restrict__ gm, float* __restrict__ gd, int N) {
    int n = blockIdx.x * blockDim.x + threadIdx.x;
    if (n >= N) return;
    int b = batch[n];
    float a = __expf(gate[n] - gm[b]);
    gate[n] = a;
    atomicAdd(&gd[b], a);
}

__global__ void pool_agg(const float* __restrict__ gate, const float* __restrict__ gd,
                         const int* __restrict__ batch, const float* __restrict__ h,
                         float* __restrict__ pooled, int N) {
    long i = (long)blockIdx.x * blockDim.x + threadIdx.x;
    if (i >= (long)N * CDIM) return;
    int n = (int)(i >> 6), c = (int)(i & 63);
    int b = batch[n];
    float w = gate[n] / gd[b];
    atomicAdd(&pooled[(size_t)b * CDIM + c], w * h[i]);
}

__global__ void final_out(const float* __restrict__ pooled, const float* __restrict__ oW,
                          const float* __restrict__ ob, float* __restrict__ out) {
    int b = threadIdx.x;
    if (b >= N_GRAPHS) return;
    const float* pp = pooled + (size_t)b * CDIM;
    float acc = ob[0];
#pragma unroll
    for (int c = 0; c < CDIM; ++c) acc += pp[c] * oW[c];
    out[b] = 1.0f / (1.0f + __expf(-acc));
}

// ---------------------------------------------------------------------------
// Host orchestration
// ---------------------------------------------------------------------------
extern "C" void kernel_launch(void* const* d_in, const int* in_sizes, int n_in,
                              void* d_out, int out_size, void* d_ws, size_t ws_size,
                              hipStream_t stream) {
    (void)in_sizes; (void)n_in; (void)out_size; (void)ws_size;

    const float* x         = (const float*)d_in[0];
    const float* edge_attr = (const float*)d_in[1];
    const int*   esrc      = (const int*)d_in[2];
    const int*   edst      = (const int*)d_in[3];
    const int*   batch     = (const int*)d_in[4];
    const float* node_W    = (const float*)d_in[5];
    const float* node_b    = (const float*)d_in[6];
    const float* edge_W    = (const float*)d_in[7];
    const float* edge_b    = (const float*)d_in[8];
    const float* Wq        = (const float*)d_in[9];
    const float* bq        = (const float*)d_in[10];
    const float* Wk        = (const float*)d_in[11];
    const float* bk        = (const float*)d_in[12];
    const float* Wv        = (const float*)d_in[13];
    const float* bv        = (const float*)d_in[14];
    const float* We        = (const float*)d_in[15];
    const float* be        = (const float*)d_in[16];
    const float* Wskip     = (const float*)d_in[17];
    const float* bskip     = (const float*)d_in[18];
    const float* gamma     = (const float*)d_in[19];
    const float* beta      = (const float*)d_in[20];
    const float* gate_W    = (const float*)d_in[21];
    const float* gate_b    = (const float*)d_in[22];
    const float* out_W     = (const float*)d_in[23];
    const float* out_b     = (const float*)d_in[24];

    const int N = N_NODES, E = N_EDGES;

    // Workspace layout (floats)
    float* ws = (float*)d_ws;
    float* h      = ws; ws += (size_t)N * CDIM;
    float* ee     = ws; ws += (size_t)E * CDIM;
    float* q      = ws; ws += (size_t)N * HC;
    float* k      = ws; ws += (size_t)N * HC;
    float* v      = ws; ws += (size_t)N * HC;
    float* ep     = ws; ws += (size_t)E * HC;
    float* lg     = ws; ws += (size_t)E * 2;
    float* m      = ws; ws += (size_t)N * 2;
    float* dn     = ws; ws += (size_t)N * 2;
    float* nxt    = ws; ws += (size_t)N * CDIM;
    float* bnsum  = ws; ws += 64;
    float* bnsq   = ws; ws += 64;
    float* mu     = ws; ws += 64;
    float* rstd   = ws; ws += 64;
    float* gate   = ws; ws += N;
    float* gm     = ws; ws += 64;
    float* gd     = ws; ws += 64;
    float* pooled = ws; ws += (size_t)N_GRAPHS * CDIM;

    const float NEG_INF = -3.402823466e38f;
    const int nodeTiles = (N + 63) / 64;   // 782
    const int edgeTiles = (E + 63) / 64;   // 8000

    // Encoders
    gemm_bias_wmma<32, 64><<<nodeTiles, 128, 0, stream>>>(x, node_W, node_b, h, N, 16);
    gemm_bias_wmma<32, 64><<<edgeTiles, 128, 0, stream>>>(edge_attr, edge_W, edge_b, ee, E, 8);

    for (int i = 0; i < 2; ++i) {
        const float* Wq_i = Wq + (size_t)i * CDIM * HC;
        const float* Wk_i = Wk + (size_t)i * CDIM * HC;
        const float* Wv_i = Wv + (size_t)i * CDIM * HC;
        const float* We_i = We + (size_t)i * CDIM * HC;
        const float* Ws_i = Wskip + (size_t)i * CDIM * CDIM;

        gemm_bias_wmma<64, 128><<<nodeTiles, 256, 0, stream>>>(h,  Wq_i, bq + i * HC, q,  N, 64);
        gemm_bias_wmma<64, 128><<<nodeTiles, 256, 0, stream>>>(h,  Wk_i, bk + i * HC, k,  N, 64);
        gemm_bias_wmma<64, 128><<<nodeTiles, 256, 0, stream>>>(h,  Wv_i, bv + i * HC, v,  N, 64);
        gemm_bias_wmma<64, 128><<<edgeTiles, 256, 0, stream>>>(ee, We_i, be + i * HC, ep, E, 64);
        gemm_bias_wmma<64, 64><<<nodeTiles, 128, 0, stream>>>(h, Ws_i, bskip + i * CDIM, nxt, N, 64);

        fill_kernel<<<(N * 2 + 255) / 256, 256, 0, stream>>>(m,  NEG_INF, (long)N * 2);
        fill_kernel<<<(N * 2 + 255) / 256, 256, 0, stream>>>(dn, 0.0f,    (long)N * 2);

        edge_logits<<<(E + 255) / 256, 256, 0, stream>>>(q, k, ep, esrc, edst, lg, E);
        seg_max   <<<(E * 2 + 255) / 256, 256, 0, stream>>>(lg, edst, m, E);
        seg_expsum<<<(E * 2 + 255) / 256, 256, 0, stream>>>(lg, edst, m, dn, E);
        edge_agg  <<<(int)(((long)E * HC + 255) / 256), 256, 0, stream>>>(
            lg, dn, v, ep, esrc, edst, nxt, E);

        fill_kernel<<<1, 128, 0, stream>>>(bnsum, 0.0f, 128);  // bnsum + bnsq contiguous
        bn_stats<<<512, 256, 0, stream>>>(nxt, bnsum, bnsq, N);
        bn_finalize<<<1, 64, 0, stream>>>(bnsum, bnsq, mu, rstd, 1.0f / (float)N);
        bn_apply<<<(int)(((long)N * CDIM + 255) / 256), 256, 0, stream>>>(
            h, nxt, mu, rstd, gamma + i * CDIM, beta + i * CDIM, N);
    }

    // Global attention pooling
    gate_kernel<<<(N + 255) / 256, 256, 0, stream>>>(h, gate_W, gate_b, gate, N);
    fill_kernel<<<1, 64, 0, stream>>>(gm, NEG_INF, 64);
    fill_kernel<<<1, 64, 0, stream>>>(gd, 0.0f, 64);
    fill_kernel<<<(N_GRAPHS * CDIM + 255) / 256, 256, 0, stream>>>(pooled, 0.0f, (long)N_GRAPHS * CDIM);
    pool_max   <<<(N + 255) / 256, 256, 0, stream>>>(gate, batch, gm, N);
    pool_expsum<<<(N + 255) / 256, 256, 0, stream>>>(gate, batch, gm, gd, N);
    pool_agg   <<<(int)(((long)N * CDIM + 255) / 256), 256, 0, stream>>>(
        gate, gd, batch, h, pooled, N);
    final_out<<<1, 64, 0, stream>>>(pooled, out_W, out_b, (float*)d_out);
}
